// NodeModel_7395933684252
// MI455X (gfx1250) — matrix operand order repaired and checked by reading, not compiled, not run on hardware
//
#include <hip/hip_runtime.h>

// ---------------------------------------------------------------------------
// Types
// ---------------------------------------------------------------------------
typedef __attribute__((ext_vector_type(16))) __bf16 v16bf;
typedef __attribute__((ext_vector_type(8)))  __bf16 v8bf;
typedef __attribute__((ext_vector_type(4)))  __bf16 v4bf;
typedef __attribute__((ext_vector_type(8)))  float  v8f;
typedef __attribute__((ext_vector_type(4)))  float  v4f;

struct Seg {
    const void*  ptr;    // float* or __bf16* source matrix [Msrc, K] row-major
    const int*   idx;    // optional row gather (length M), nullptr = identity
    const float* mean;   // optional per-col norm (length K), nullptr = identity
    const float* rstd;
    int          K;      // columns of this segment (multiple of 32)
    int          bf16;   // 1 if ptr is __bf16
};

#define BM 64
#define BN 64
#define BK 32
#define A_STRIDE 40   // halves: 80B row stride; 8B/16B aligned chunks
#define B_STRIDE 48   // halves: 96B row stride; 32B aligned chunks

// ---------------------------------------------------------------------------
// One input segment's K-range, software-pipelined (reg prefetch + double-
// buffered LDS, one barrier per K-step). Called once per segment with the
// segment's fields as SEPARATE scalar args -> everything stays in SGPRs,
// no indexable aggregate, no scratch.
// ---------------------------------------------------------------------------
__device__ __forceinline__ void gemm_segment(
    const void* __restrict__ sptr, const int* __restrict__ sidxp,
    const float* __restrict__ smean, const float* __restrict__ srstd,
    int sK, int sbf, int kwbase,
    const float* __restrict__ Wm, int N, int nbase, int M, int mbase,
    __bf16 (&sA)[2][BM][A_STRIDE], __bf16 (&sBT)[2][BN][B_STRIDE],
    v8f (&acc)[4])
{
    if (sK <= 0) return;                       // uniform across block

    const int tid    = threadIdx.x;
    const int wave   = tid >> 5;
    const int lane   = tid & 31;
    const int l16    = lane & 15;
    const int hiHalf = lane >> 4;
    // A-stage geometry: 4 rows (arow0 + 16*i) x one 4-wide column chunk
    const int arow0 = tid >> 3;
    const int ac4   = (tid & 7) * 4;
    // W-stage geometry: one n column, 16 consecutive k values
    const int wn  = tid & 63;
    const int wkh = (tid >> 6) * 16;

    bool vld[4]; int srow[4];
#pragma unroll
    for (int i = 0; i < 4; ++i) {
        int gm = mbase + i * 16 + arow0;
        vld[i] = gm < M;
        int g  = vld[i] ? gm : 0;
        srow[i] = sidxp ? sidxp[g] : g;
    }

    // prefetch registers
    v4f  paf[4]; v4bf pab[4];
    v4f  pmn = {}, prs = {};
    float pw[16];

    auto prefetch = [&](int kk) {
#pragma unroll
        for (int i = 0; i < 4; ++i) {
            if (vld[i]) {
                long long o = (long long)srow[i] * sK + kk + ac4;
                if (sbf) pab[i] = *(const v4bf*)((const __bf16*)sptr + o);
                else     paf[i] = *(const v4f*)((const float*)sptr + o);
            }
        }
        if (smean) {
            pmn = *(const v4f*)&smean[kk + ac4];
            prs = *(const v4f*)&srstd[kk + ac4];
        }
#pragma unroll
        for (int j = 0; j < 16; ++j)
            pw[j] = Wm[(long long)(kwbase + kk + wkh + j) * N + nbase + wn];
    };

    auto commit = [&](int buf) {
#pragma unroll
        for (int i = 0; i < 4; ++i) {
            float x0 = 0.f, x1 = 0.f, x2 = 0.f, x3 = 0.f;
            if (vld[i]) {
                if (sbf) {
                    x0 = (float)pab[i][0]; x1 = (float)pab[i][1];
                    x2 = (float)pab[i][2]; x3 = (float)pab[i][3];
                } else {
                    x0 = paf[i][0]; x1 = paf[i][1];
                    x2 = paf[i][2]; x3 = paf[i][3];
                }
                if (smean) {
                    x0 = (x0 - pmn[0]) * prs[0]; x1 = (x1 - pmn[1]) * prs[1];
                    x2 = (x2 - pmn[2]) * prs[2]; x3 = (x3 - pmn[3]) * prs[3];
                }
            }
            v4bf ov;
            ov[0] = (__bf16)x0; ov[1] = (__bf16)x1;
            ov[2] = (__bf16)x2; ov[3] = (__bf16)x3;
            *(v4bf*)&sA[buf][i * 16 + arow0][ac4] = ov;        // 8B ds_store
        }
#pragma unroll
        for (int j = 0; j < 16; j += 4) {
            v4bf ov;
            ov[0] = (__bf16)pw[j];     ov[1] = (__bf16)pw[j + 1];
            ov[2] = (__bf16)pw[j + 2]; ov[3] = (__bf16)pw[j + 3];
            *(v4bf*)&sBT[buf][wn][wkh + j] = ov;               // 8B ds_store
        }
    };

    auto compute = [&](int buf) {
        const int nrow = wave * 16 + l16;
        v8bf blo = *(const v8bf*)&sBT[buf][nrow][hiHalf * 8];
        v8bf bhi = *(const v8bf*)&sBT[buf][nrow][16 + hiHalf * 8];
        v16bf b = __builtin_shufflevector(blo, bhi,
                  0,1,2,3,4,5,6,7,8,9,10,11,12,13,14,15);
        const int klo = hiHalf * 8;        // A: interleaved K-halves per ISA
#pragma unroll
        for (int ms = 0; ms < 4; ++ms) {
            const int arow = ms * 16 + l16;
            v8bf alo = *(const v8bf*)&sA[buf][arow][klo];
            v8bf ahi = *(const v8bf*)&sA[buf][arow][16 + klo];
            v16bf a = __builtin_shufflevector(alo, ahi,
                      0,1,2,3,4,5,6,7,8,9,10,11,12,13,14,15);
            acc[ms] = __builtin_amdgcn_wmma_f32_16x16x32_bf16(
                          false, a, false, b, (short)0, acc[ms], false, false);
        }
    };

    // pipelined loop over this segment's K-range
    const int T = sK / BK;
    prefetch(0);
    commit(0);
    __syncthreads();
    for (int t = 0; t < T; ++t) {
        const int cur = t & 1;
        if (t + 1 < T) prefetch((t + 1) * BK);  // issue global loads
        compute(cur);                           // WMMA hides load latency
        if (t + 1 < T) commit(cur ^ 1);         // wait + store to other buffer
        __syncthreads();
    }
}

// ---------------------------------------------------------------------------
// WMMA GEMM:  out[M,N] = relu( concat(seg0,seg1,seg2)[M,Ktot] @ W[Ktot,N] + b )
// epilogue accumulates per-column sum / sumsq (valid rows only) via atomics.
// Block = 128 threads (4 waves). Tile: 64(M) x 64(N), K-step 32 (bf16 WMMA).
// ---------------------------------------------------------------------------
__global__ __launch_bounds__(128)
void gemm_bias_relu_stats(Seg s0, Seg s1, Seg s2,
                          const float* __restrict__ Wm,
                          const float* __restrict__ bias,
                          __bf16* __restrict__ out,
                          float* __restrict__ ssum,
                          float* __restrict__ ssq,
                          int M, int N)
{
    __shared__ __attribute__((aligned(32))) __bf16 sA [2][BM][A_STRIDE]; // [m][k]
    __shared__ __attribute__((aligned(32))) __bf16 sBT[2][BN][B_STRIDE]; // [n][k]

    const int tid    = threadIdx.x;
    const int wave   = tid >> 5;
    const int lane   = tid & 31;
    const int l16    = lane & 15;
    const int hiHalf = lane >> 4;
    const int mbase  = blockIdx.x * BM;
    const int nbase  = blockIdx.y * BN;

    v8f acc[4] = {};

    // three explicit calls -> compile-time-constant segment field selection
    gemm_segment(s0.ptr, s0.idx, s0.mean, s0.rstd, s0.K, s0.bf16,
                 0,             Wm, N, nbase, M, mbase, sA, sBT, acc);
    gemm_segment(s1.ptr, s1.idx, s1.mean, s1.rstd, s1.K, s1.bf16,
                 s0.K,          Wm, N, nbase, M, mbase, sA, sBT, acc);
    gemm_segment(s2.ptr, s2.idx, s2.mean, s2.rstd, s2.K, s2.bf16,
                 s0.K + s1.K,   Wm, N, nbase, M, mbase, sA, sBT, acc);

    // ---- epilogue: bias + relu + bf16 store + column stats ----
    const int   n  = nbase + wave * 16 + l16;
    const float bn = bias[n];
    float accS = 0.0f, accQ = 0.0f;
#pragma unroll
    for (int ms = 0; ms < 4; ++ms) {
#pragma unroll
        for (int r = 0; r < 8; ++r) {
            int m = mbase + ms * 16 + hiHalf * 8 + r;   // D layout per ISA
            float v = acc[ms][r] + bn;
            v = v > 0.0f ? v : 0.0f;
            if (m < M) {
                out[(long long)m * N + n] = (__bf16)v;
                accS += v; accQ += v * v;
            }
        }
    }
    atomicAdd(&ssum[n], accS);
    atomicAdd(&ssq[n],  accQ);
}

// ---------------------------------------------------------------------------
// Small helper kernels
// ---------------------------------------------------------------------------
__global__ __launch_bounds__(256)
void zero_k(float* p, int n) {
    int i = blockIdx.x * 256 + threadIdx.x;
    if (i < n) p[i] = 0.0f;
}

__global__ __launch_bounds__(256)
void finalize_k(const float* __restrict__ sum, const float* __restrict__ sq,
                float* __restrict__ mean, float* __restrict__ rstd,
                int N, float M) {
    int i = blockIdx.x * 256 + threadIdx.x;
    if (i < N) {
        float mu  = sum[i] / M;
        float var = (sq[i] - sum[i] * mu) / (M - 1.0f);   // unbiased (ddof=1)
        mean[i] = mu;
        rstd[i] = rsqrtf(var + 1e-5f);
    }
}

__global__ __launch_bounds__(256)
void count_k(const int* __restrict__ idx, float* __restrict__ cnt, int E) {
    int e = blockIdx.x * 256 + threadIdx.x;
    if (e < E) atomicAdd(&cnt[idx[e]], 1.0f);
}

// one block per edge row, 256 threads = D; applies norm of x on the fly
__global__ __launch_bounds__(256)
void scatter_k(const __bf16* __restrict__ x,
               const float* __restrict__ mean, const float* __restrict__ rstd,
               const int* __restrict__ idx, float* __restrict__ acc, int E) {
    int e = blockIdx.x;
    int d = threadIdx.x;
    float v = (float)x[(long long)e * 256 + d];
    v = (v - mean[d]) * rstd[d];
    atomicAdd(&acc[(long long)idx[e] * 256 + d], v);
}

__global__ __launch_bounds__(256)
void divide_k(float* __restrict__ acc, const float* __restrict__ cnt, int n) {
    int i = blockIdx.x;
    int d = threadIdx.x;
    float c = cnt[i]; c = c > 1.0f ? c : 1.0f;
    acc[(long long)i * 256 + d] /= c;
}

// final gnorm: bf16 pre-norm -> f32 d_out
__global__ __launch_bounds__(256)
void normalize_out_k(const __bf16* __restrict__ x,
                     const float* __restrict__ mean, const float* __restrict__ rstd,
                     float* __restrict__ out, int M) {
    int i = blockIdx.x;
    int d = threadIdx.x;
    float v = (float)x[(long long)i * 256 + d];
    out[(long long)i * 256 + d] = (v - mean[d]) * rstd[d];
}

// ---------------------------------------------------------------------------
// Host orchestration
// ---------------------------------------------------------------------------
static const Seg SEG_NULL = {nullptr, nullptr, nullptr, nullptr, 0, 0};

static void run_gemm(hipStream_t st, Seg s0, Seg s1, Seg s2,
                     const float* W, const float* b, __bf16* out,
                     float* ssum, float* ssq, int M, int N) {
    dim3 grid((M + BM - 1) / BM, N / BN);
    gemm_bias_relu_stats<<<grid, 128, 0, st>>>(s0, s1, s2, W, b, out, ssum, ssq, M, N);
}

static void zero(hipStream_t st, float* p, int n) {
    zero_k<<<(n + 255) / 256, 256, 0, st>>>(p, n);
}

// stats slice layout per GEMM stage: [sum(512) | sumsq(512) | mean(512) | rstd(512)]
static float* st_ptr(float* stats, int stage, int which) {
    return stats + (size_t)stage * 2048 + (size_t)which * 512;
}

// full mlp: hid = relu(in@w1+b1) (pre-norm, stats stage s), out = relu(gn(hid)@w2+b2)
// (pre-norm, stats stage s+1). Consumers read `out` with stage s+1 mean/rstd.
static void run_mlp(hipStream_t st, Seg s0, Seg s1, Seg s2,
                    const float* w1, const float* b1,
                    const float* w2, const float* b2,
                    int M, __bf16* hid, __bf16* out, float* stats, int stage) {
    float *sum0 = st_ptr(stats, stage, 0),     *sq0 = st_ptr(stats, stage, 1);
    float *mn0  = st_ptr(stats, stage, 2),     *rs0 = st_ptr(stats, stage, 3);
    float *sum1 = st_ptr(stats, stage + 1, 0), *sq1 = st_ptr(stats, stage + 1, 1);
    float *mn1  = st_ptr(stats, stage + 1, 2), *rs1 = st_ptr(stats, stage + 1, 3);

    zero(st, sum0, 1024);
    run_gemm(st, s0, s1, s2, w1, b1, hid, sum0, sq0, M, 512);
    finalize_k<<<2, 256, 0, st>>>(sum0, sq0, mn0, rs0, 512, (float)M);

    zero(st, sum1, 1024);
    Seg sh = {hid, nullptr, mn0, rs0, 512, 1};
    run_gemm(st, sh, SEG_NULL, SEG_NULL, w2, b2, out, sum1, sq1, M, 256);
    finalize_k<<<1, 256, 0, st>>>(sum1, sq1, mn1, rs1, 256, (float)M);
}

static void run_smean(hipStream_t st, const __bf16* x, const float* mean,
                      const float* rstd, const int* idx, int E, int n,
                      float* agg, float* cnt) {
    zero(st, agg, n * 256);
    zero(st, cnt, n);
    count_k<<<(E + 255) / 256, 256, 0, st>>>(idx, cnt, E);
    scatter_k<<<E, 256, 0, st>>>(x, mean, rstd, idx, agg, E);
    divide_k<<<n, 256, 0, st>>>(agg, cnt, n);
}

extern "C" void kernel_launch(void* const* d_in, const int* in_sizes, int n_in,
                              void* d_out, int out_size, void* d_ws, size_t ws_size,
                              hipStream_t stream) {
    const int N_INS = 10000, N_LBL = 1000, E_INS = 160000, E_CROSS = 80000;

    const float* node_ins        = (const float*)d_in[0];
    const float* edge_attr_ins   = (const float*)d_in[1];
    const float* node_label      = (const float*)d_in[2];
    const float* edge_attr_cross = (const float*)d_in[3];
    const int*   row_ins         = (const int*)d_in[4];
    const int*   col_ins         = (const int*)d_in[5];
    const int*   row_cross       = (const int*)d_in[6];
    const int*   col_cross       = (const int*)d_in[7];

    // Detect param dict flattening order via the unique 768x512 ins.w1.
    int g_enc, g_mlp1, g_mlp2, g_inner, g_inter, g_ins, g_li, g_lbl;
    if (n_in >= 40 && in_sizes[8 + 4 * 2] == 768 * 512) {
        // alphabetical: enc, inner, ins, inter, lbl, lbl_inter, mlp1, mlp2
        g_enc = 0; g_inner = 1; g_ins = 2; g_inter = 3;
        g_lbl = 4; g_li = 5; g_mlp1 = 6; g_mlp2 = 7;
    } else {
        // insertion: enc, mlp1, mlp2, inner, inter, ins, lbl_inter, lbl
        g_enc = 0; g_mlp1 = 1; g_mlp2 = 2; g_inner = 3;
        g_inter = 4; g_ins = 5; g_li = 6; g_lbl = 7;
    }
    auto P = [&](int g, int j) { return (const float*)d_in[8 + 4 * g + j]; };

    // ---- workspace carve-out (~277 MB) ----
    char* ws = (char*)d_ws;
    size_t off = 0;
    auto carve = [&](size_t bytes) {
        void* p = ws + off; off += (bytes + 255) & ~(size_t)255; return p;
    };
    __bf16* HID  = (__bf16*)carve((size_t)E_INS * 512 * 2);   // layer-1 pre-norm
    __bf16* EOUT = (__bf16*)carve((size_t)E_INS * 256 * 2);   // layer-2 pre-norm (edges)
    __bf16* HA   = (__bf16*)carve((size_t)N_INS * 256 * 2);
    __bf16* HB   = (__bf16*)carve((size_t)N_INS * 256 * 2);
    float*  AGG1 = (float*)carve((size_t)N_INS * 256 * 4);
    float*  AGG2 = (float*)carve((size_t)N_INS * 256 * 4);
    float*  CNT  = (float*)carve((size_t)N_INS * 4);
    float*  STATS= (float*)carve((size_t)16 * 2048 * 4);
    (void)ws_size;

    float* out_h   = (float*)d_out;                       // [10000,256]
    float* out_lbl = (float*)d_out + (size_t)N_INS * 256; // [1000,256]
    (void)out_size;

    auto MN = [&](int stage) { return st_ptr(STATS, stage, 2); };
    auto RS = [&](int stage) { return st_ptr(STATS, stage, 3); };

    // 1) h0 = mlp(node_ins, enc)                         stages 0,1 -> HA
    Seg sin = {node_ins, nullptr, nullptr, nullptr, 128, 0};
    run_mlp(stream, sin, SEG_NULL, SEG_NULL,
            P(g_enc,0), P(g_enc,1), P(g_enc,2), P(g_enc,3),
            N_INS, HID, HA, STATS, 0);

    // 2) a = mlp([h0[row_ins], edge_attr_ins], mlp1)     stages 2,3 -> EOUT
    Seg sh0g = {HA, row_ins, MN(1), RS(1), 256, 1};
    Seg sei  = {edge_attr_ins, nullptr, nullptr, nullptr, 256, 0};
    run_mlp(stream, sh0g, sei, SEG_NULL,
            P(g_mlp1,0), P(g_mlp1,1), P(g_mlp1,2), P(g_mlp1,3),
            E_INS, HID, EOUT, STATS, 2);

    // 3) smean(a, col_ins) -> AGG1
    run_smean(stream, EOUT, MN(3), RS(3), col_ins, E_INS, N_INS, AGG1, CNT);

    // 4) h1 = mlp([h0, agg], mlp2)                       stages 4,5 -> HB
    Seg sh0 = {HA, nullptr, MN(1), RS(1), 256, 1};
    Seg sa1 = {AGG1, nullptr, nullptr, nullptr, 256, 0};
    run_mlp(stream, sh0, sa1, SEG_NULL,
            P(g_mlp2,0), P(g_mlp2,1), P(g_mlp2,2), P(g_mlp2,3),
            N_INS, HID, HB, STATS, 4);

    // 5) inner = smean(mlp([h1[row_ins], edge_attr_ins], inner), col_ins)  stages 6,7
    Seg sh1g = {HB, row_ins, MN(5), RS(5), 256, 1};
    run_mlp(stream, sh1g, sei, SEG_NULL,
            P(g_inner,0), P(g_inner,1), P(g_inner,2), P(g_inner,3),
            E_INS, HID, EOUT, STATS, 6);
    run_smean(stream, EOUT, MN(7), RS(7), col_ins, E_INS, N_INS, AGG1, CNT);

    // 6) inter = smean(mlp([label[col_cross], edge_attr_cross], inter), row_cross)  8,9
    Seg slg = {node_label, col_cross, nullptr, nullptr, 256, 0};
    Seg sec = {edge_attr_cross, nullptr, nullptr, nullptr, 256, 0};
    run_mlp(stream, slg, sec, SEG_NULL,
            P(g_inter,0), P(g_inter,1), P(g_inter,2), P(g_inter,3),
            E_CROSS, HID, EOUT, STATS, 8);
    run_smean(stream, EOUT, MN(9), RS(9), row_cross, E_CROSS, N_INS, AGG2, CNT);

    // 7) h = mlp([h1, inner, inter], ins)                stages 10,11 -> HA -> out_h
    Seg sh1 = {HB, nullptr, MN(5), RS(5), 256, 1};
    Seg sg1 = {AGG1, nullptr, nullptr, nullptr, 256, 0};
    Seg sg2 = {AGG2, nullptr, nullptr, nullptr, 256, 0};
    run_mlp(stream, sh1, sg1, sg2,
            P(g_ins,0), P(g_ins,1), P(g_ins,2), P(g_ins,3),
            N_INS, HID, HA, STATS, 10);
    normalize_out_k<<<N_INS, 256, 0, stream>>>(HA, MN(11), RS(11), out_h, N_INS);

    // 8) l = smean(mlp([h[row_cross], edge_attr_cross], lbl_inter), col_cross)  12,13
    Seg shg = {out_h, row_cross, nullptr, nullptr, 256, 0};  // normalized h, f32
    run_mlp(stream, shg, sec, SEG_NULL,
            P(g_li,0), P(g_li,1), P(g_li,2), P(g_li,3),
            E_CROSS, HID, EOUT, STATS, 12);
    run_smean(stream, EOUT, MN(13), RS(13), col_cross, E_CROSS, N_LBL, AGG1, CNT);

    // 9) lbl = mlp([node_label, l], lbl)                 stages 14,15 -> HB -> out_lbl
    Seg snl = {node_label, nullptr, nullptr, nullptr, 256, 0};
    Seg sl  = {AGG1, nullptr, nullptr, nullptr, 256, 0};
    run_mlp(stream, snl, sl, SEG_NULL,
            P(g_lbl,0), P(g_lbl,1), P(g_lbl,2), P(g_lbl,3),
            N_LBL, HID, HB, STATS, 14);
    normalize_out_k<<<N_LBL, 256, 0, stream>>>(HB, MN(15), RS(15), out_lbl, N_LBL);
}
